// SubMConv3d_35931696398506
// MI455X (gfx1250) — matrix operand order repaired and checked by reading, not compiled, not run on hardware
//
#include <hip/hip_runtime.h>
#include <hip/hip_bf16.h>

// ---- CDNA5 WMMA fragment types (wave32) ----
typedef __attribute__((ext_vector_type(16))) __bf16 v16bf;
typedef __attribute__((ext_vector_type(8)))  float  v8f;

#define HASH_CAP  (1u << 20)          // 1M entries, load factor ~0.38
#define HASH_MASK (HASH_CAP - 1u)
#define EMPTY_ENTRY 0xFFFFFFFFFFFFFFFFull
#define DSX 256
#define DSY 256
#define DSZ 32
#define CIN 32
#define COUT 32
#define NTAPS 27
#define WFRAGS (NTAPS * 2 * 32)             // v16bf fragments: tap * col-half * lane
#define WFRAG_ELEMS (WFRAGS * 16)           // bf16 elements

// ------------------------------------------------------------------
// Kernel 1: clear hash table
// ------------------------------------------------------------------
__global__ void init_hash_kernel(unsigned long long* __restrict__ tab) {
    unsigned i = blockIdx.x * blockDim.x + threadIdx.x;
    if (i < HASH_CAP) tab[i] = EMPTY_ENTRY;
}

// ------------------------------------------------------------------
// Kernel 2: insert (key -> point index) with 64-bit CAS, linear probe
// key = ((b*SX + x)*SY + y)*SZ + z  < 2^23, so entry never aliases EMPTY
// ------------------------------------------------------------------
__global__ void build_hash_kernel(const long long* __restrict__ ind,
                                  unsigned long long* __restrict__ tab, int n) {
    int i = blockIdx.x * blockDim.x + threadIdx.x;
    if (i >= n) return;
    unsigned b = (unsigned)ind[i * 4 + 0];
    unsigned x = (unsigned)ind[i * 4 + 1];
    unsigned y = (unsigned)ind[i * 4 + 2];
    unsigned z = (unsigned)ind[i * 4 + 3];
    unsigned key = ((b * DSX + x) * DSY + y) * DSZ + z;
    unsigned long long entry = ((unsigned long long)key << 32) | (unsigned)i;
    unsigned h = (key * 2654435761u) & HASH_MASK;
    while (atomicCAS(&tab[h], EMPTY_ENTRY, entry) != EMPTY_ENTRY)
        h = (h + 1) & HASH_MASK;
}

// ------------------------------------------------------------------
// Kernel 3: repack weights (COUT,CIN,3,3,3) fp32 -> bf16 B-fragments.
// B is 32x16 (KxN) per (tap, col-half). ISA 16-bit B layout:
//   lane half h: v[e] (e<8) = B[8h+e][lane%16], v[8+e] = B[16+8h+e][lane%16]
// Flat index t = ((tap*2+half)*32 + lane)*16 + e
// ------------------------------------------------------------------
__global__ void prep_weights_kernel(const float* __restrict__ w,
                                    __bf16* __restrict__ wfrag) {
    int t = blockIdx.x * blockDim.x + threadIdx.x;
    if (t >= WFRAG_ELEMS) return;
    int e    = t & 15;
    int lane = (t >> 4) & 31;
    int h    = (t >> 9) & 1;
    int k    = t >> 10;
    int cin  = (e >> 3) * 16 + (lane >> 4) * 8 + (e & 7);        // K index
    int cout = h * 16 + (lane & 15);        // N index
    int dx = k / 9, dy = (k / 3) % 3, dz = k % 3;
    float v = w[(((cout * CIN + cin) * 3 + dx) * 3 + dy) * 3 + dz];
    wfrag[t] = (__bf16)v;
}

// ------------------------------------------------------------------
// Hash lookup (read-only, table is L2-resident: 8MB << 192MB L2)
// ------------------------------------------------------------------
__device__ __forceinline__ int hash_lookup(const unsigned long long* __restrict__ tab,
                                           unsigned key) {
    unsigned h = (key * 2654435761u) & HASH_MASK;
    for (;;) {
        unsigned long long e = tab[h];
        if (e == EMPTY_ENTRY) return -1;
        if ((unsigned)(e >> 32) == key) return (int)(e & 0xFFFFFFFFu);
        h = (h + 1) & HASH_MASK;
    }
}

// ------------------------------------------------------------------
// Kernel 4: main gather + WMMA kernel.
// 8 waves/block, each wave owns 16 output rows.
// - B-fragment table (55 KB) staged once per block into LDS; per-tap B
//   reads become ds_load_b128 (short dscnt wait) instead of L2 round-trips.
// - Nested dx/dy/dz loops: tap offsets are compile-time constants, no
//   SALU divide chains; compiler may fully unroll into 27 straight-line
//   tap bodies, each ending in two v_wmma_f32_16x16x32_bf16.
// - Wave-uniform __any skip: at ~4.8% density most taps have zero matches.
// - EXEC is full at every WMMA (divergence is lane-predicated selects only;
//   branches around WMMA are wave-uniform).
// ------------------------------------------------------------------
__global__ __launch_bounds__(256) void subm_wmma_kernel(
    const float* __restrict__ feat, const long long* __restrict__ ind,
    const float* __restrict__ bias, const unsigned long long* __restrict__ tab,
    const v16bf* __restrict__ wfrag, float* __restrict__ out, int n) {

    __shared__ v16bf lds_w[WFRAGS];        // 27*2*32 * 32B = 55296 B

    // ---- cooperative stage of all B fragments into LDS ----
    {
        const uint4* src = (const uint4*)wfrag;
        uint4*       dst = (uint4*)lds_w;
        for (int i = threadIdx.x; i < WFRAGS * 2; i += 256) dst[i] = src[i];
    }
    __syncthreads();

    int wave = threadIdx.x >> 5;
    int lane = threadIdx.x & 31;
    int base = (blockIdx.x * 8 + wave) * 16;
    if (base >= n) return;                 // wave-uniform

    int  m        = lane & 15;
    int  hi       = lane >> 4;
    int  row      = base + m;
    bool rowvalid = row < n;
    int  crow     = rowvalid ? row : (n - 1);

    int b = (int)ind[crow * 4 + 0];
    int x = (int)ind[crow * 4 + 1];
    int y = (int)ind[crow * 4 + 2];
    int z = (int)ind[crow * 4 + 3];

    v8f acc0 = {};
    v8f acc1 = {};

    int k = 0;
    for (int dx = -1; dx <= 1; ++dx)
    for (int dy = -1; dy <= 1; ++dy)
    for (int dz = -1; dz <= 1; ++dz, ++k) {
        int nbr;
        if (dx == 0 && dy == 0 && dz == 0) {   // center tap: always self
            nbr = rowvalid ? row : -1;
        } else {
            int nx = x + dx, ny = y + dy, nz = z + dz;
            bool inb = rowvalid && (unsigned)nx < DSX && (unsigned)ny < DSY &&
                       (unsigned)nz < DSZ;
            nbr = inb ? hash_lookup(tab, ((unsigned)((b * DSX + nx) * DSY + ny)) * DSZ
                                             + (unsigned)nz)
                      : -1;
        }
        if (!__any(nbr >= 0)) continue;    // wave-uniform skip, EXEC stays full

        // Gather A fragment: row = tile point m, K range selected by hi.
        // ISA 16-bit A 16x32 layout: v[e] (e<8) -> K=hi*8+e ; v[8+e] -> K=16+hi*8+e
        v16bf a;
        if (nbr >= 0) {
            const float4* f4 = (const float4*)(feat + (size_t)nbr * CIN);
            float4 u0 = f4[hi * 2];
            float4 u1 = f4[hi * 2 + 1];
            float4 w0 = f4[4 + hi * 2];
            float4 w1 = f4[4 + hi * 2 + 1];
            a[0]  = (__bf16)u0.x; a[1]  = (__bf16)u0.y; a[2]  = (__bf16)u0.z; a[3]  = (__bf16)u0.w;
            a[4]  = (__bf16)u1.x; a[5]  = (__bf16)u1.y; a[6]  = (__bf16)u1.z; a[7]  = (__bf16)u1.w;
            a[8]  = (__bf16)w0.x; a[9]  = (__bf16)w0.y; a[10] = (__bf16)w0.z; a[11] = (__bf16)w0.w;
            a[12] = (__bf16)w1.x; a[13] = (__bf16)w1.y; a[14] = (__bf16)w1.z; a[15] = (__bf16)w1.w;
        } else {
#pragma unroll
            for (int e = 0; e < 16; ++e) a[e] = (__bf16)0.0f;
        }

        v16bf b0 = lds_w[(k * 2 + 0) * 32 + lane];   // cout 0-15  (ds_load_b128 x2)
        v16bf b1 = lds_w[(k * 2 + 1) * 32 + lane];   // cout 16-31

        acc0 = __builtin_amdgcn_wmma_f32_16x16x32_bf16(false, a, false, b0,
                                                       (short)0, acc0, false, false);
        acc1 = __builtin_amdgcn_wmma_f32_16x16x32_bf16(false, a, false, b1,
                                                       (short)0, acc1, false, false);
    }

    // C/D layout: VGPR r, lane -> M = r + 8*hi, N = m. Fuse bias.
    float bias0 = bias[m];
    float bias1 = bias[16 + m];
#pragma unroll
    for (int r = 0; r < 8; ++r) {
        int prow = base + r + 8 * hi;
        if (prow < n) {
            out[prow * COUT + m]      = acc0[r] + bias0;
            out[prow * COUT + 16 + m] = acc1[r] + bias1;
        }
    }
}

// ------------------------------------------------------------------
extern "C" void kernel_launch(void* const* d_in, const int* in_sizes, int n_in,
                              void* d_out, int out_size, void* d_ws, size_t ws_size,
                              hipStream_t stream) {
    const float*     feat   = (const float*)d_in[0];      // (N, 32) f32
    const long long* ind    = (const long long*)d_in[1];  // (N, 4)  i64
    const float*     weight = (const float*)d_in[2];      // (32,32,3,3,3) f32
    const float*     bias   = (const float*)d_in[3];      // (32,)   f32
    float*           out    = (float*)d_out;              // (N, 32) f32
    int n = in_sizes[0] / CIN;

    unsigned long long* tab = (unsigned long long*)d_ws;               // 8 MB
    __bf16* wfrag = (__bf16*)((char*)d_ws + (size_t)HASH_CAP * 8);     // 54 KB

    hipLaunchKernelGGL(init_hash_kernel, dim3((HASH_CAP + 255) / 256), dim3(256),
                       0, stream, tab);
    hipLaunchKernelGGL(build_hash_kernel, dim3((n + 255) / 256), dim3(256),
                       0, stream, ind, tab, n);
    hipLaunchKernelGGL(prep_weights_kernel, dim3((WFRAG_ELEMS + 255) / 256), dim3(256),
                       0, stream, weight, wfrag);
    hipLaunchKernelGGL(subm_wmma_kernel, dim3((n + 127) / 128), dim3(256),
                       0, stream, feat, ind, bias, tab, (const v16bf*)wfrag, out, n);
}
